// CachedAttention_15882789060758
// MI455X (gfx1250) — matrix operand order repaired and checked by reading, not compiled, not run on hardware
//
#include <hip/hip_runtime.h>
#include <hip/hip_bf16.h>

typedef __attribute__((ext_vector_type(16))) _Float16 v16h;
typedef __attribute__((ext_vector_type(8)))  _Float16 v8h;
typedef __attribute__((ext_vector_type(8)))  float    v8f;

static constexpr int Bc  = 4;
static constexpr int Tc  = 2048;
static constexpr int Dc  = 2048;
static constexpr int Hc  = 16;
static constexpr int DKc = 128;
static constexpr int Mc  = Bc * Tc;     // 8192 rows

// ---- CDNA5 feature guards (fall back cleanly if builtin names differ) ----
#if defined(__gfx1250__) && __has_builtin(__builtin_amdgcn_global_load_async_to_lds_b128)
#define USE_ASYNC_LDS 1
#else
#define USE_ASYNC_LDS 0
#endif
#if defined(__gfx1250__) && __has_builtin(__builtin_amdgcn_tensor_load_to_lds) && \
    __has_builtin(__builtin_amdgcn_s_wait_tensorcnt)
#define USE_TDM 1
#else
#define USE_TDM 0
#endif

#if USE_ASYNC_LDS
// Builtin signature (from hipcc diagnostic): param 1 is a pointer to
// 'int __attribute__((vector_size(16)))' in the global (device) addr space.
typedef int v4i_vs __attribute__((vector_size(16)));
typedef __attribute__((address_space(1))) v4i_vs as1_v4i;
typedef __attribute__((address_space(3))) v4i_vs as3_v4i;
static __device__ __forceinline__ as1_v4i* to_glob(const void* p) {
  return (as1_v4i*)(unsigned long long)p;            // generic->AS1 is identity
}
static __device__ __forceinline__ as3_v4i* to_lds(void* p) {
  return (as3_v4i*)(unsigned)(unsigned long long)p;  // flat LDS addr: low 32 bits
}
#endif

// ---------------- helpers ----------------
static __device__ __forceinline__ v16h hcat(v8h a, v8h b) {
  return __builtin_shufflevector(a, b, 0,1,2,3,4,5,6,7,8,9,10,11,12,13,14,15);
}
static __device__ __forceinline__ v8f fzero() {
  v8f z; for (int i = 0; i < 8; ++i) z[i] = 0.0f; return z;
}

#if USE_TDM
typedef __attribute__((ext_vector_type(4))) unsigned int u32x4;
typedef __attribute__((ext_vector_type(8))) int i32x8;
typedef __attribute__((ext_vector_type(4))) int i32x4;

// Issue a 2D tile DMA (global -> LDS) via the Tensor Data Mover.
// D# bitfields per CDNA5 ISA ch.10 "Tensor DMA Descriptor":
//   group0: count=1 | lds_addr | global_addr(57b) | type=2
//   group1: data_size=2B, pad_enable, pad_interval/amount, dims/strides/tile
static __device__ __forceinline__ void tdm_load_2d(
    const void* gaddr, void* laddr,
    unsigned td0, unsigned td1, unsigned long long stride0,
    unsigned tile0, unsigned tile1, unsigned padi, unsigned pada) {
  unsigned long long ga = (unsigned long long)gaddr;
  u32x4 g0;
  g0[0] = 1u;                                              // count=1 (valid user D#)
  g0[1] = (unsigned)(unsigned long long)laddr;             // lds_addr (bytes)
  g0[2] = (unsigned)ga;                                    // global_addr[31:0]
  g0[3] = (unsigned)((ga >> 32) & 0x01FFFFFFu) | (2u << 30); // ga[56:32] | type=2
  i32x8 g1;
  g1[0] = (int)((1u << 16) | (1u << 20) | (padi << 22) | (pada << 25)); // 2B elems, pad
  g1[1] = (int)((td0 & 0xFFFFu) << 16);                    // tensor_dim0[15:0]
  g1[2] = (int)((td0 >> 16) | ((td1 & 0xFFFFu) << 16));    // dim0 hi | dim1 lo
  g1[3] = (int)((td1 >> 16) | (tile0 << 16));              // dim1 hi | tile_dim0
  g1[4] = (int)tile1;                                      // tile_dim1 (tile_dim2=0)
  g1[5] = (int)(unsigned)stride0;                          // tensor_dim0_stride lo
  g1[6] = (int)((stride0 >> 32) & 0xFFFFu);                // stride hi (dim1_stride=0)
  g1[7] = 0;
  i32x4 z4 = {0, 0, 0, 0};
#if __clang_major__ >= 23
  i32x8 z8 = {0, 0, 0, 0, 0, 0, 0, 0};
  __builtin_amdgcn_tensor_load_to_lds(g0, g1, z4, z4, z8, 0);
#else
  __builtin_amdgcn_tensor_load_to_lds(g0, g1, z4, z4, 0);
#endif
}
#endif

// ---------------- fp32 -> f16 convert ----------------
__global__ void ca_cvt_f32_f16(const float* __restrict__ s, _Float16* __restrict__ d, int n) {
  int i = (blockIdx.x * blockDim.x + threadIdx.x) * 4;
  if (i + 3 < n) {
    float4 v = *(const float4*)(s + i);
    d[i+0] = (_Float16)v.x; d[i+1] = (_Float16)v.y;
    d[i+2] = (_Float16)v.z; d[i+3] = (_Float16)v.w;
  } else {
    for (; i < n; ++i) d[i] = (_Float16)s[i];
  }
}

// ---------------- GEMM: C[M,N] = A[M,K] * W[N,K]^T ----------------
// MODE 0: f16 out, head-major scatter  [B][H][T][DK]   (Q and K)
// MODE 1: f16 out, transposed scatter  [B][H][DK][T]   (V)
// MODE 2: f32 out, row-major [M][D]                    (final output)
static constexpr int BM = 128, BN = 128, BK = 32;
static constexpr int LDT  = BK + 8;     // LDS row stride (halves)
static constexpr int TILE = BM * LDT;   // one buffer

template <int MODE>
__global__ __launch_bounds__(256)
void ca_gemm_bt(const _Float16* __restrict__ A, const _Float16* __restrict__ W,
                void* __restrict__ dstv) {
  __shared__ _Float16 As[2 * TILE];     // ping-pong A tiles
  __shared__ _Float16 Bs[2 * TILE];     // ping-pong B tiles

  const int m0   = blockIdx.y * BM;
  const int n0   = blockIdx.x * BN;
  const int tid  = threadIdx.x;
  const int lane = tid & 31;
  const int w    = tid >> 5;     // 8 waves
  const int wm   = w & 3;        // 4 M sub-tiles of 32 rows
  const int wn   = w >> 2;       // 2 N sub-tiles of 64 cols
  const int hl   = lane >> 4;    // half-wave select
  const int l16  = lane & 15;

  const int row = tid >> 1, seg = (tid & 1) * 16;
  auto stage = [&](int buf, int kk) {
    const _Float16* sa = A + (size_t)(m0 + row) * Dc + kk + seg;
    _Float16*       da = &As[buf * TILE + row * LDT + seg];
    const _Float16* sb = W + (size_t)(n0 + row) * Dc + kk + seg;
    _Float16*       db = &Bs[buf * TILE + row * LDT + seg];
#if USE_ASYNC_LDS
    __builtin_amdgcn_global_load_async_to_lds_b128(to_glob(sa),     to_lds(da),     0, 0);
    __builtin_amdgcn_global_load_async_to_lds_b128(to_glob(sa + 8), to_lds(da + 8), 0, 0);
    __builtin_amdgcn_global_load_async_to_lds_b128(to_glob(sb),     to_lds(db),     0, 0);
    __builtin_amdgcn_global_load_async_to_lds_b128(to_glob(sb + 8), to_lds(db + 8), 0, 0);
#else
    ((v8h*)da)[0] = ((const v8h*)sa)[0]; ((v8h*)da)[1] = ((const v8h*)sa)[1];
    ((v8h*)db)[0] = ((const v8h*)sb)[0]; ((v8h*)db)[1] = ((const v8h*)sb)[1];
#endif
  };

  v8f acc[2][4];
  for (int i = 0; i < 2; ++i)
    for (int j = 0; j < 4; ++j) acc[i][j] = fzero();

  const int NK = Dc / BK;
  stage(0, 0);                                   // prefetch first tile
  for (int it = 0; it < NK; ++it) {
    const int buf = it & 1;
#if USE_ASYNC_LDS
    asm volatile("s_wait_asynccnt 0x0" ::: "memory");  // my DMAs for tile `it` done
#endif
    __syncthreads();                             // tile `it` visible; buf^1 readers retired
    if (it + 1 < NK) stage(buf ^ 1, (it + 1) * BK);    // overlap next-tile DMA w/ compute

    const _Float16* Ab = &As[buf * TILE];
    const _Float16* Bb = &Bs[buf * TILE];
    v16h afr[2];
    for (int mf = 0; mf < 2; ++mf) {
      const _Float16* p = &Ab[(wm * 32 + mf * 16 + l16) * LDT];
      int klo = hl * 8;
      afr[mf] = hcat(*(const v8h*)(p + klo), *(const v8h*)(p + 16 + klo));
    }
    for (int nf = 0; nf < 4; ++nf) {
      const _Float16* p = &Bb[(wn * 64 + nf * 16 + l16) * LDT];
      int kb = hl * 8;
      v16h bfr = hcat(*(const v8h*)(p + kb), *(const v8h*)(p + kb + 16));
      for (int mf = 0; mf < 2; ++mf)
        acc[mf][nf] = __builtin_amdgcn_wmma_f32_16x16x32_f16(
            false, afr[mf], false, bfr, (short)0, acc[mf][nf], false, false);
    }
  }

  for (int mf = 0; mf < 2; ++mf)
    for (int nf = 0; nf < 4; ++nf)
      for (int r = 0; r < 8; ++r) {
        int gm = m0 + wm * 32 + mf * 16 + hl * 8 + r;
        int gn = n0 + wn * 64 + nf * 16 + l16;
        float v = acc[mf][nf][r];
        if (MODE == 2) {
          ((float*)dstv)[(size_t)gm * Dc + gn] = v;
        } else {
          int b = gm / Tc, t = gm % Tc;
          int h = gn / DKc, dk = gn % DKc;
          size_t idx = (MODE == 0)
            ? (((size_t)(b * Hc + h) * Tc + t) * DKc + dk)     // [B][H][T][DK]
            : (((size_t)(b * Hc + h) * DKc + dk) * Tc + t);    // [B][H][DK][T]
          ((_Float16*)dstv)[idx] = (_Float16)v;
        }
      }
}

// ---------------- causal flash attention ----------------
// grid: (T/64, B*H); block: 128 threads = 4 waves; each wave owns 16 query rows.
__global__ __launch_bounds__(128)
void ca_flash(const _Float16* __restrict__ Qh, const _Float16* __restrict__ Kh,
              const _Float16* __restrict__ Vt, _Float16* __restrict__ AO) {
  constexpr int LK = DKc + 8;   // K tile stride: 64 data DW + 4 pad DW per row
  constexpr int LV = 32 + 8;    // V^T tile stride: 16 data DW + 4 pad DW per row
  constexpr int LP = 32 + 8;    // P tile stride
  constexpr int KT = 32 * LK;   // one K buffer
  constexpr int VT = DKc * LV;  // one V buffer
  __shared__ _Float16 Ks[2 * KT];        // ping-pong 32 keys x 128 dk
  __shared__ _Float16 Vs[2 * VT];        // ping-pong 128 dk x 32 keys (V^T)
  __shared__ _Float16 Ps[4 * 16 * LP];   // per-wave 16 x 32 probability tile

  const int tid  = threadIdx.x;
  const int lane = tid & 31;
  const int w    = tid >> 5;
  const int hl   = lane >> 4;
  const int l16  = lane & 15;

  const int bh = blockIdx.y;
  const int b  = bh / Hc, h = bh % Hc;
  const int q0 = blockIdx.x * 64 + w * 16;

  const _Float16* qp = Qh + (size_t)bh * Tc * DKc;
  const _Float16* kp = Kh + (size_t)bh * Tc * DKc;
  const _Float16* vp = Vt + (size_t)bh * DKc * Tc;

  auto issue = [&](int buf, int kb) {
#if USE_TDM
    if (w == 0) {
      // K tile: 32(keys) rows x 128(dk), row stride DK; 64+4 DWORDs/row in LDS
      tdm_load_2d(kp + (size_t)kb * DKc, &Ks[buf * KT], DKc, Tc, DKc, DKc, 32, 5u, 3u);
      // V^T tile: 128(dk) rows x 32(keys), row stride T; 16+4 DWORDs/row in LDS
      tdm_load_2d(vp + kb, &Vs[buf * VT], Tc, DKc, Tc, 32, DKc, 3u, 3u);
    }
#else
    int row = tid >> 2, seg = (tid & 3) * 32;
    const v8h* s = (const v8h*)(kp + (size_t)(kb + row) * DKc + seg);
    v8h* d = (v8h*)&Ks[buf * KT + row * LK + seg];
    d[0] = s[0]; d[1] = s[1]; d[2] = s[2]; d[3] = s[3];
    const v8h* s2 = (const v8h*)(vp + (size_t)tid * Tc + kb);
    v8h* d2 = (v8h*)&Vs[buf * VT + tid * LV];
    d2[0] = s2[0]; d2[1] = s2[1]; d2[2] = s2[2]; d2[3] = s2[3];
#endif
  };

  // Q fragments: 4 K-chunks of 32 over DK=128
  v16h aq[4];
  for (int c = 0; c < 4; ++c) {
    const _Float16* p = qp + (size_t)(q0 + l16) * DKc + c * 32 + hl * 8;
    aq[c] = hcat(*(const v8h*)p, *(const v8h*)(p + 16));
  }

  v8f o[8];
  for (int nt = 0; nt < 8; ++nt) o[nt] = fzero();
  float mst[8], lst[8];
  for (int r = 0; r < 8; ++r) { mst[r] = -INFINITY; lst[r] = 0.0f; }

  const float scale = 0.08838834764831845f;  // 1/sqrt(128)
  const int ktiles = blockIdx.x * 2 + 2;     // keys [0, blockIdx.x*64 + 64)

  issue(0, 0);                               // prefetch first K/V tile
  for (int kt = 0; kt < ktiles; ++kt) {
    const int kb = kt * 32, buf = kt & 1;
#if USE_TDM
    if (w == 0) __builtin_amdgcn_s_wait_tensorcnt(0);  // DMAs for tile `kt` done
#endif
    __syncthreads();                         // tile visible; buf^1 readers retired
    if (kt + 1 < ktiles) issue(buf ^ 1, kb + 32);      // overlap next DMA w/ compute

    const _Float16* Kb = &Ks[buf * KT];
    const _Float16* Vb = &Vs[buf * VT];

    // S = Q * K^T  for 32 keys (two 16-key WMMA column tiles)
    v8f s0 = fzero(), s1 = fzero();
    for (int c = 0; c < 4; ++c) {
      const _Float16* p0 = &Kb[l16 * LK + c * 32 + hl * 8];
      v16h bk0 = hcat(*(const v8h*)p0, *(const v8h*)(p0 + 16));
      s0 = __builtin_amdgcn_wmma_f32_16x16x32_f16(false, aq[c], false, bk0,
                                                  (short)0, s0, false, false);
      const _Float16* p1 = &Kb[(16 + l16) * LK + c * 32 + hl * 8];
      v16h bk1 = hcat(*(const v8h*)p1, *(const v8h*)(p1 + 16));
      s1 = __builtin_amdgcn_wmma_f32_16x16x32_f16(false, aq[c], false, bk1,
                                                  (short)0, s1, false, false);
    }

    // online softmax over this 32-key slab
    float pr0[8], pr1[8];
    for (int r = 0; r < 8; ++r) {
      int qi = q0 + hl * 8 + r;
      float x0 = (kb + l16      <= qi) ? s0[r] * scale : -1e30f;
      float x1 = (kb + 16 + l16 <= qi) ? s1[r] * scale : -1e30f;
      float mx = fmaxf(x0, x1);
      mx = fmaxf(mx, __shfl_xor(mx, 1));
      mx = fmaxf(mx, __shfl_xor(mx, 2));
      mx = fmaxf(mx, __shfl_xor(mx, 4));
      mx = fmaxf(mx, __shfl_xor(mx, 8));
      float mnew = fmaxf(mst[r], mx);
      float p0 = __expf(x0 - mnew);
      float p1 = __expf(x1 - mnew);
      float sum = p0 + p1;
      sum += __shfl_xor(sum, 1);
      sum += __shfl_xor(sum, 2);
      sum += __shfl_xor(sum, 4);
      sum += __shfl_xor(sum, 8);
      float alpha = __expf(mst[r] - mnew);
      lst[r] = lst[r] * alpha + sum;
      mst[r] = mnew;
      for (int nt = 0; nt < 8; ++nt) o[nt][r] *= alpha;
      pr0[r] = p0; pr1[r] = p1;
    }

    // C-layout -> A-layout transpose of P via wave-private LDS tile
    _Float16* pw = &Ps[w * 16 * LP];
    for (int r = 0; r < 8; ++r) {
      int m = hl * 8 + r;
      pw[m * LP + l16]      = (_Float16)pr0[r];
      pw[m * LP + 16 + l16] = (_Float16)pr1[r];
    }
    asm volatile("s_wait_dscnt 0" ::: "memory");

    const _Float16* prd = &Ps[(w * 16 + l16) * LP];
    int klo = hl * 8;
    v16h ap = hcat(*(const v8h*)(prd + klo), *(const v8h*)(prd + 16 + klo));

    // O += P * V  (8 output column tiles over DK=128)
    for (int nt = 0; nt < 8; ++nt) {
      const _Float16* pv = &Vb[(nt * 16 + l16) * LV + hl * 8];
      v16h bv = hcat(*(const v8h*)pv, *(const v8h*)(pv + 16));
      o[nt] = __builtin_amdgcn_wmma_f32_16x16x32_f16(false, ap, false, bv,
                                                     (short)0, o[nt], false, false);
    }
  }

  // normalize and merge heads into [B][T][D] f16
  for (int nt = 0; nt < 8; ++nt)
    for (int r = 0; r < 8; ++r) {
      int t   = q0 + hl * 8 + r;
      int gdk = nt * 16 + l16;
      float val = o[nt][r] / lst[r];
      AO[(size_t)(b * Tc + t) * Dc + h * DKc + gdk] = (_Float16)val;
    }
}

// ---------------- host side ----------------
extern "C" void kernel_launch(void* const* d_in, const int* in_sizes, int n_in,
                              void* d_out, int out_size, void* d_ws, size_t ws_size,
                              hipStream_t stream) {
  const float* x  = (const float*)d_in[0];
  const float* Wq = (const float*)d_in[1];
  const float* Wk = (const float*)d_in[2];
  const float* Wv = (const float*)d_in[3];
  const float* Wo = (const float*)d_in[4];
  (void)in_sizes; (void)n_in; (void)out_size; (void)ws_size;

  char* ws = (char*)d_ws;
  const size_t SZ_X = (size_t)Mc * Dc * 2;       // 32 MB
  const size_t SZ_W = (size_t)Dc * Dc * 2;       //  8 MB
  _Float16* xh  = (_Float16*)(ws);
  _Float16* Wqh = (_Float16*)(ws + SZ_X);
  _Float16* Wkh = (_Float16*)(ws + SZ_X + SZ_W);
  _Float16* Wvh = (_Float16*)(ws + SZ_X + 2 * SZ_W);
  _Float16* Woh = (_Float16*)(ws + SZ_X + 3 * SZ_W);
  _Float16* Qh  = (_Float16*)(ws + SZ_X + 4 * SZ_W);
  _Float16* Kh  = (_Float16*)(ws + 2 * SZ_X + 4 * SZ_W);
  _Float16* Vt  = (_Float16*)(ws + 3 * SZ_X + 4 * SZ_W);
  _Float16* AOh = (_Float16*)(ws + 4 * SZ_X + 4 * SZ_W);

  const int NX = Mc * Dc;       // 16.7M
  const int NW = Dc * Dc;       //  4.2M
  ca_cvt_f32_f16<<<NX / 1024, 256, 0, stream>>>(x,  xh,  NX);
  ca_cvt_f32_f16<<<NW / 1024, 256, 0, stream>>>(Wq, Wqh, NW);
  ca_cvt_f32_f16<<<NW / 1024, 256, 0, stream>>>(Wk, Wkh, NW);
  ca_cvt_f32_f16<<<NW / 1024, 256, 0, stream>>>(Wv, Wvh, NW);
  ca_cvt_f32_f16<<<NW / 1024, 256, 0, stream>>>(Wo, Woh, NW);

  dim3 gg(Dc / BN, Mc / BM);    // (16, 64)
  ca_gemm_bt<0><<<gg, 256, 0, stream>>>(xh, Wqh, (void*)Qh);
  ca_gemm_bt<0><<<gg, 256, 0, stream>>>(xh, Wkh, (void*)Kh);
  ca_gemm_bt<1><<<gg, 256, 0, stream>>>(xh, Wvh, (void*)Vt);

  ca_flash<<<dim3(Tc / 64, Bc * Hc), 128, 0, stream>>>(Qh, Kh, Vt, AOh);

  ca_gemm_bt<2><<<gg, 256, 0, stream>>>(AOh, Woh, d_out);
}
